// Volume_Rendering_40759239639178
// MI455X (gfx1250) — matrix operand order repaired and checked by reading, not compiled, not run on hardware
//
#include <hip/hip_runtime.h>
#include <hip/hip_bf16.h>
#include <math.h>

typedef __attribute__((ext_vector_type(16))) _Float16 v16h;
typedef __attribute__((ext_vector_type(8)))  float    v8f;

#define BB 2
#define RES 64
#define M_RAYS (RES*RES)        // 4096 rays per batch
#define DC 48
#define DFN 48
#define CF 32
#define PR 256
#define PTOT (BB*M_RAYS*DC)     // 393216 points per pass
#define NRAY (BB*M_RAYS)        // 8192 rays

__device__ __forceinline__ float rnd01(unsigned s){
    s ^= s >> 16; s *= 0x7feb352dU; s ^= s >> 15; s *= 0x846ca68bU; s ^= s >> 16;
    return (float)(s >> 8) * (1.0f/16777216.0f);
}
// branchless, numerically stable, hardware-transcendental softplus:
// softplus(x) = max(x,0) + log(1 + exp(-|x|));  exp(-|x|) <= 1 so no overflow.
__device__ __forceinline__ float softplus_f(float x){
    return fmaxf(x, 0.f) + __logf(1.f + __expf(-fabsf(x)));
}
__device__ __forceinline__ float sigmoid_f(float x){
    return __fdividef(1.0f, 1.0f + __expf(-x));
}

// ---------------- weight prep: scaled f16, [n][k] layout ----------------
__global__ void k_prep(const float* __restrict__ W1, const float* __restrict__ b1,
                       const float* __restrict__ W2, const float* __restrict__ b2,
                       _Float16* __restrict__ W1h, _Float16* __restrict__ W2h,
                       float* __restrict__ b1f, float* __restrict__ b2f){
    int n = threadIdx.x;                 // 0..63
    const float g1 = 0.17677669529663687f;   // 1/sqrt(32)
    const float g2 = 0.125f;                 // 1/sqrt(64)
    if (n < 64){
        b1f[n] = b1[n];
        for (int k = 0; k < 32; ++k) W1h[n*32 + k] = (_Float16)(W1[n*32 + k] * g1);
    }
    if (n < 48){
        b2f[n] = (n < 33) ? b2[n] : 0.f;
        for (int k = 0; k < 64; ++k)
            W2h[n*64 + k] = (_Float16)((n < 33) ? W2[n*64 + k] * g2 : 0.f);
    }
}

// ---------------- ray sampler ----------------
__global__ void k_rays(const float* __restrict__ c, float* __restrict__ ro,
                       float* __restrict__ rd){
    int idx = blockIdx.x*blockDim.x + threadIdx.x;
    if (idx >= NRAY) return;
    int n = idx / M_RAYS, m = idx % M_RAYS;
    const float* cn = c + n*25;
    float fx = cn[16], sk = cn[17], cx = cn[18];
    float fy = cn[20], cy = cn[21];
    int ii = m / RES, jj = m % RES;
    float x_cam = (jj + 0.5f) / RES;
    float y_cam = (ii + 0.5f) / RES;
    float x_lift = (x_cam - cx + cy*sk/fy - sk*y_cam/fy) / fx;
    float y_lift = (y_cam - cy) / fy;
    float p0 = x_lift, p1 = y_lift, p2 = 1.f, p3 = 1.f;
    float w[3];
    #pragma unroll
    for (int r = 0; r < 3; ++r)
        w[r] = cn[r*4+0]*p0 + cn[r*4+1]*p1 + cn[r*4+2]*p2 + cn[r*4+3]*p3;
    float ox = cn[3], oy = cn[7], oz = cn[11];
    float dx = w[0]-ox, dy = w[1]-oy, dz = w[2]-oz;
    float nrm = sqrtf(dx*dx + dy*dy + dz*dz);
    float inv = 1.0f / fmaxf(nrm, 1e-12f);
    ro[idx*3+0] = ox; ro[idx*3+1] = oy; ro[idx*3+2] = oz;
    rd[idx*3+0] = dx*inv; rd[idx*3+1] = dy*inv; rd[idx*3+2] = dz*inv;
}

// ---------------- stratified coarse depths ----------------
__global__ void k_dcoarse(float* __restrict__ dc){
    int p = blockIdx.x*blockDim.x + threadIdx.x;
    if (p >= PTOT) return;
    int d = p % DC;
    const float delta = (3.3f - 2.25f) / (DC - 1);
    dc[p] = 2.25f + d*delta + rnd01((unsigned)p*2654435761u + 12345u)*delta;
}

// ---------------- tri-plane bilinear sampling ----------------
__device__ __forceinline__ float tap(const float* __restrict__ fp,
                                     const float* __restrict__ cp, int x, int y){
    if (x < 0 || x >= PR || y < 0 || y >= PR) return 0.f;
    int o = y*PR + x;
    return fp[o] + cp[o];
}

__global__ void k_sample(const float* __restrict__ planes, const float* __restrict__ canon,
                         const float* __restrict__ ro, const float* __restrict__ rd,
                         const float* __restrict__ depths, _Float16* __restrict__ feats){
    long long t = (long long)blockIdx.x*blockDim.x + threadIdx.x;
    if (t >= (long long)PTOT*CF) return;
    int ch = (int)(t & 31);
    int p  = (int)(t >> 5);
    int n  = p / (M_RAYS*DC);
    int r  = p % (M_RAYS*DC);
    int m  = r / DC;
    int ray = n*M_RAYS + m;
    float dep = depths[p];
    float c0 = (ro[ray*3+0] + dep*rd[ray*3+0]) * 2.0f;   // 2/BOX_WARP
    float c1 = (ro[ray*3+1] + dep*rd[ray*3+1]) * 2.0f;
    float c2 = (ro[ray*3+2] + dep*rd[ray*3+2]) * 2.0f;
    float acc = 0.f;
    #pragma unroll
    for (int pl = 0; pl < 3; ++pl){
        float gx, gy;
        if      (pl == 0){ gx = c0; gy = c1; }
        else if (pl == 1){ gx = c0; gy = c2; }
        else             { gx = c2; gy = c0; }
        float x = (gx + 1.f)*(PR*0.5f) - 0.5f;
        float y = (gy + 1.f)*(PR*0.5f) - 0.5f;
        float x0f = floorf(x), y0f = floorf(y);
        float tx = x - x0f, ty = y - y0f;
        int x0 = (int)x0f, y0 = (int)y0f;
        const float* fp = planes + ((long long)(n*3 + pl)*CF + ch)*(PR*PR);
        const float* cp = canon  + ((long long)pl*CF + ch)*(PR*PR);
        float v00 = tap(fp, cp, x0,   y0  );
        float v10 = tap(fp, cp, x0+1, y0  );
        float v01 = tap(fp, cp, x0,   y0+1);
        float v11 = tap(fp, cp, x0+1, y0+1);
        acc += v00*(1-tx)*(1-ty) + v10*tx*(1-ty) + v01*(1-tx)*ty + v11*tx*ty;
    }
    feats[(long long)p*CF + ch] = (_Float16)(acc * (1.0f/3.0f));
}

// ---------------- WMMA MLP decode ----------------
union F16x16 { v16h h; unsigned u[8]; uint4 q[2]; };

__global__ void __launch_bounds__(256) k_mlp(const _Float16* __restrict__ feats,
        const _Float16* __restrict__ W1h, const _Float16* __restrict__ W2h,
        const float* __restrict__ b1f, const float* __restrict__ b2f,
        _Float16* __restrict__ colors, float* __restrict__ sigma){
    __shared__ _Float16 lds[8][16][64];
    int lane = threadIdx.x & 31;
    int wave = threadIdx.x >> 5;
    int tile = blockIdx.x*8 + wave;
    int base = tile*16;
    int mlo  = lane & 15;
    int hi   = lane >> 4;

    // ---- layer 1: A = X tile (16x32 f16), ISA A-layout ----
    // lanes 0-15: VGPR0-3 K=0..7, VGPR4-7 K=16..23; lanes 16-31: +8
    F16x16 a1;
    {
        const uint4* fp = (const uint4*)(feats + (long long)(base + mlo)*32);
        a1.q[0] = fp[hi];       // halves 8*hi .. 8*hi+7
        a1.q[1] = fp[2 + hi];   // halves 16+8*hi .. 16+8*hi+7
    }
    v8f acc[4];
    #pragma unroll
    for (int j = 0; j < 4; ++j){
        int nn = j*16 + mlo;
        F16x16 b;
        const uint4* wp = (const uint4*)(W1h + nn*32);
        b.q[0] = wp[hi]; b.q[1] = wp[2 + hi];   // K = 8*hi..+7, 16+8*hi..+7 for column nn
        v8f z = {0,0,0,0,0,0,0,0};
        acc[j] = __builtin_amdgcn_wmma_f32_16x16x32_f16(
                     false, a1.h, false, b.h, (short)0, z, false, false);
    }
    // bias + softplus, stage H through LDS (D-layout -> A-layout transpose)
    #pragma unroll
    for (int j = 0; j < 4; ++j){
        #pragma unroll
        for (int rr = 0; rr < 8; ++rr){
            int mm = rr + 8*hi;
            int nn = j*16 + mlo;
            lds[wave][mm][nn] = (_Float16)softplus_f(acc[j][rr] + b1f[nn]);
        }
    }
    __syncthreads();

    // ---- layer 2: A = H tile (16x64 f16) as two K=32 fragments ----
    F16x16 a2lo, a2hi;
    {
        const uint4* hp = (const uint4*)&lds[wave][mlo][0];
        a2lo.q[0] = hp[hi];       a2lo.q[1] = hp[2 + hi];   // K 0..31
        a2hi.q[0] = hp[4 + hi];   a2hi.q[1] = hp[6 + hi];   // K 32..63
    }
    #pragma unroll
    for (int j = 0; j < 3; ++j){
        int nn = j*16 + mlo;
        F16x16 blo, bhi;
        const uint4* wp = (const uint4*)(W2h + nn*64);
        blo.q[0] = wp[hi];     blo.q[1] = wp[2 + hi];   // K = 8*hi..+7, 16+8*hi..+7
        bhi.q[0] = wp[4 + hi]; bhi.q[1] = wp[6 + hi];   // K = 32+8*hi..+7, 48+8*hi..+7
        v8f z = {0,0,0,0,0,0,0,0};
        v8f o = __builtin_amdgcn_wmma_f32_16x16x32_f16(
                    false, a2lo.h, false, blo.h, (short)0, z, false, false);
        o = __builtin_amdgcn_wmma_f32_16x16x32_f16(
                    false, a2hi.h, false, bhi.h, (short)0, o, false, false);
        #pragma unroll
        for (int rr = 0; rr < 8; ++rr){
            int mm = rr + 8*hi;
            int pt = base + mm;
            float val = o[rr] + b2f[nn];
            if (nn == 0)      sigma[pt] = val;
            else if (nn < 33) colors[(long long)pt*32 + (nn-1)] =
                                  (_Float16)(sigmoid_f(val)*1.002f - 0.001f);
        }
    }
}

// ---------------- coarse ray marcher (weights only) ----------------
__global__ void k_march_coarse(const float* __restrict__ sigma,
                               const float* __restrict__ dc, float* __restrict__ wout){
    int ray = blockIdx.x*blockDim.x + threadIdx.x;
    if (ray >= NRAY) return;
    const float* s = sigma + (long long)ray*DC;
    const float* d = dc    + (long long)ray*DC;
    float T = 1.f;
    for (int i = 0; i < DC-1; ++i){
        float delta = d[i+1] - d[i];
        float dm    = softplus_f(0.5f*(s[i] + s[i+1]) - 1.0f);
        float alpha = 1.f - __expf(-dm*delta);
        wout[(long long)ray*(DC-1) + i] = alpha*T;
        T *= (1.f - alpha + 1e-10f);
    }
}

// ---------------- importance sampling (inverse CDF) ----------------
__global__ void k_importance(const float* __restrict__ dc, const float* __restrict__ wc,
                             float* __restrict__ dfout){
    int ray = blockIdx.x*blockDim.x + threadIdx.x;
    if (ray >= NRAY) return;
    float d[DC], w[DC-1];
    for (int i = 0; i < DC;   ++i) d[i] = dc[(long long)ray*DC + i];
    for (int i = 0; i < DC-1; ++i) w[i] = wc[(long long)ray*(DC-1) + i];
    float wmax[DC];
    wmax[0] = w[0]; wmax[DC-1] = w[DC-2];
    for (int i = 1; i < DC-1; ++i) wmax[i] = fmaxf(w[i-1], w[i]);
    float w2[DC-1], mids[DC-1];
    for (int i = 0; i < DC-1; ++i){
        w2[i]   = 0.5f*(wmax[i] + wmax[i+1]) + 0.01f;
        mids[i] = 0.5f*(d[i] + d[i+1]);
    }
    const int NS = DC - 3;              // 45 interior weights
    float cdf[DC-2];                    // NS+1 = 46 entries
    cdf[0] = 0.f;
    float sum = 0.f;
    for (int i = 0; i < NS; ++i) sum += w2[1+i] + 1e-5f;
    float run = 0.f;
    float rsum = __fdividef(1.f, sum);
    for (int i = 0; i < NS; ++i){ run += (w2[1+i] + 1e-5f)*rsum; cdf[i+1] = run; }
    for (int s = 0; s < DFN; ++s){
        float u = rnd01((unsigned)(ray*DFN + s)*0x9E3779B9u + 0xABCD1234u);
        int ind = 0;
        for (int i = 0; i <= NS; ++i) if (cdf[i] <= u) ind = i + 1;   // side='right'
        int below = ind - 1 > 0 ? ind - 1 : 0;
        int above = ind < NS ? ind : NS;
        float cl = cdf[below], ch = cdf[above];
        float bl = mids[below], bh = mids[above];
        float den = ch - cl; if (den < 1e-5f) den = 1.f;
        dfout[(long long)ray*DFN + s] = bl + __fdividef(u - cl, den)*(bh - bl);
    }
}

// ---------------- unify (stable sort) + final marcher scalars ----------------
__global__ void k_unify_march(const float* __restrict__ dc, const float* __restrict__ dfi,
                              const float* __restrict__ sig_c, const float* __restrict__ sig_f,
                              float* __restrict__ wout, int* __restrict__ sidx,
                              float* __restrict__ depth_out){
    int ray = blockIdx.x*blockDim.x + threadIdx.x;
    if (ray >= NRAY) return;
    const int NT = DC + DFN;            // 96
    float d[NT]; int ix[NT];
    for (int i = 0; i < DC;  ++i) d[i]       = dc [(long long)ray*DC  + i];
    for (int i = 0; i < DFN; ++i) d[DC + i]  = dfi[(long long)ray*DFN + i];
    for (int i = 0; i < NT;  ++i) ix[i] = i;
    for (int i = 1; i < NT; ++i){        // stable insertion sort == argsort
        float dv = d[i]; int iv = ix[i]; int j = i - 1;
        while (j >= 0 && d[j] > dv){ d[j+1] = d[j]; ix[j+1] = ix[j]; --j; }
        d[j+1] = dv; ix[j+1] = iv;
    }
    float sg[NT];
    for (int i = 0; i < NT; ++i){
        int s0 = ix[i];
        sg[i] = (s0 < DC) ? sig_c[(long long)ray*DC + s0]
                          : sig_f[(long long)ray*DFN + (s0 - DC)];
    }
    float T = 1.f, dnum = 0.f, wsum = 0.f;
    for (int i = 0; i < NT-1; ++i){
        float delta = d[i+1] - d[i];
        float dm    = softplus_f(0.5f*(sg[i] + sg[i+1]) - 1.f);
        float alpha = 1.f - __expf(-dm*delta);
        float w = alpha*T;
        T *= (1.f - alpha + 1e-10f);
        wout[(long long)ray*(NT-1) + i] = w;
        dnum += w*0.5f*(d[i] + d[i+1]);
        wsum += w;
        sidx[(long long)ray*NT + i] = ix[i];
    }
    sidx[(long long)ray*NT + (NT-1)] = ix[NT-1];
    float dep = __fdividef(dnum, wsum);
    if (!(dep == dep)) dep = d[NT-1];        // nan -> inf -> clipped to max
    dep = fminf(fmaxf(dep, d[0]), d[NT-1]);
    depth_out[ray] = dep;
}

// ---------------- per-channel rgb accumulation ----------------
__global__ void k_rgb(const _Float16* __restrict__ col_c, const _Float16* __restrict__ col_f,
                      const float* __restrict__ w95, const int* __restrict__ sidx,
                      float* __restrict__ rgbf){
    int t = blockIdx.x*blockDim.x + threadIdx.x;
    if (t >= NRAY*CF) return;
    int ch = t & 31; int ray = t >> 5;
    const int NT = DC + DFN;
    const int*   ix = sidx + (long long)ray*NT;
    const float* w  = w95  + (long long)ray*(NT-1);
    auto colAt = [&](int i) -> float {
        int s = ix[i];
        return (s < DC) ? (float)col_c[((long long)ray*DC  + s       )*CF + ch]
                        : (float)col_f[((long long)ray*DFN + (s - DC))*CF + ch];
    };
    float c0 = colAt(0), acc = 0.f;
    for (int i = 0; i < NT-1; ++i){
        float c1 = colAt(i+1);
        acc += w[i]*0.5f*(c0 + c1);
        c0 = c1;
    }
    rgbf[(long long)ray*CF + ch] = acc*2.f - 1.f;
}

// ---------------- output assembly: 2x nearest upsample ----------------
__global__ void k_out(const float* __restrict__ rgbf, const float* __restrict__ depth,
                      float* __restrict__ out){
    const int HW = 128*128;
    int t = blockIdx.x*blockDim.x + threadIdx.x;
    if (t >= BB*CF*HW) return;
    int x  = t % 128;
    int y  = (t/128) % 128;
    int ch = (t/HW) % CF;
    int n  = t/(HW*CF);
    int m  = (y >> 1)*RES + (x >> 1);
    float v = rgbf[((long long)(n*M_RAYS + m))*CF + ch];
    float* feat = out + (size_t)BB*3*HW + (size_t)BB*HW;
    feat[t] = v;
    if (ch < 3)  out[(size_t)(n*3 + ch)*HW + y*128 + x] = v;
    if (ch == 0){
        float* dp = out + (size_t)BB*3*HW;
        dp[(size_t)n*HW + y*128 + x] = depth[n*M_RAYS + m];
    }
}

// ---------------- host launcher ----------------
extern "C" void kernel_launch(void* const* d_in, const int* in_sizes, int n_in,
                              void* d_out, int out_size, void* d_ws, size_t ws_size,
                              hipStream_t stream) {
    const float* c      = (const float*)d_in[0];
    const float* planes = (const float*)d_in[1];
    const float* canon  = (const float*)d_in[2];
    const float* W1     = (const float*)d_in[3];
    const float* b1     = (const float*)d_in[4];
    const float* W2     = (const float*)d_in[5];
    const float* b2     = (const float*)d_in[6];
    float* out = (float*)d_out;

    char* ws = (char*)d_ws;
    size_t off = 0;
    auto carve = [&](size_t bytes) -> char* {
        char* p = ws + off;
        off = (off + bytes + 255) & ~(size_t)255;
        return p;
    };
    _Float16* W1h   = (_Float16*)carve(64*32*2);
    _Float16* W2h   = (_Float16*)carve(48*64*2);
    float*    b1f   = (float*)carve(64*4);
    float*    b2f   = (float*)carve(48*4);
    float*    ro    = (float*)carve((size_t)NRAY*3*4);
    float*    rd    = (float*)carve((size_t)NRAY*3*4);
    float*    dc    = (float*)carve((size_t)PTOT*4);
    float*    dfb   = (float*)carve((size_t)PTOT*4);
    _Float16* feats = (_Float16*)carve((size_t)PTOT*CF*2);
    _Float16* col_c = (_Float16*)carve((size_t)PTOT*CF*2);
    _Float16* col_f = (_Float16*)carve((size_t)PTOT*CF*2);
    float*    sig_c = (float*)carve((size_t)PTOT*4);
    float*    sig_f = (float*)carve((size_t)PTOT*4);
    float*    w_c   = (float*)carve((size_t)NRAY*(DC-1)*4);
    float*    w95   = (float*)carve((size_t)NRAY*(DC+DFN-1)*4);
    int*      sidx  = (int*)carve((size_t)NRAY*(DC+DFN)*4);
    float*    depth = (float*)carve((size_t)NRAY*4);
    float*    rgbf  = (float*)carve((size_t)NRAY*CF*4);

    hipLaunchKernelGGL(k_prep, dim3(1), dim3(64), 0, stream,
                       W1, b1, W2, b2, W1h, W2h, b1f, b2f);
    hipLaunchKernelGGL(k_rays, dim3(NRAY/256), dim3(256), 0, stream, c, ro, rd);
    hipLaunchKernelGGL(k_dcoarse, dim3(PTOT/256), dim3(256), 0, stream, dc);

    // coarse pass
    hipLaunchKernelGGL(k_sample, dim3((PTOT*CF)/256), dim3(256), 0, stream,
                       planes, canon, ro, rd, dc, feats);
    hipLaunchKernelGGL(k_mlp, dim3(PTOT/128), dim3(256), 0, stream,
                       feats, W1h, W2h, b1f, b2f, col_c, sig_c);
    hipLaunchKernelGGL(k_march_coarse, dim3(NRAY/256), dim3(256), 0, stream,
                       sig_c, dc, w_c);
    hipLaunchKernelGGL(k_importance, dim3(NRAY/256), dim3(256), 0, stream,
                       dc, w_c, dfb);

    // fine pass
    hipLaunchKernelGGL(k_sample, dim3((PTOT*CF)/256), dim3(256), 0, stream,
                       planes, canon, ro, rd, dfb, feats);
    hipLaunchKernelGGL(k_mlp, dim3(PTOT/128), dim3(256), 0, stream,
                       feats, W1h, W2h, b1f, b2f, col_f, sig_f);

    // unify + final render
    hipLaunchKernelGGL(k_unify_march, dim3(NRAY/256), dim3(256), 0, stream,
                       dc, dfb, sig_c, sig_f, w95, sidx, depth);
    hipLaunchKernelGGL(k_rgb, dim3((NRAY*CF)/256), dim3(256), 0, stream,
                       col_c, col_f, w95, sidx, rgbf);
    hipLaunchKernelGGL(k_out, dim3((BB*CF*128*128)/256), dim3(256), 0, stream,
                       rgbf, depth, out);
    (void)in_sizes; (void)n_in; (void)out_size; (void)ws_size;
}
